// MMUpdateRGB_17506286698523
// MI455X (gfx1250) — matrix operand order repaired and checked by reading, not compiled, not run on hardware
//
#include <hip/hip_runtime.h>
#include <stdint.h>

// ---------------- types for WMMA (gfx1250, wave32) ----------------
typedef __attribute__((ext_vector_type(16))) __bf16          v16bf;
typedef __attribute__((ext_vector_type(8)))  __bf16          bf16x8;
typedef __attribute__((ext_vector_type(8)))  float           v8f;
typedef __attribute__((ext_vector_type(8)))  unsigned short  us8;

#define NP   784      // 28*28 query patches
#define DQ   1152     // query feature dim
#define MBK  32768    // memory bank rows
#define EMB  2304     // 2*DQ

__device__ __forceinline__ unsigned short f2bf(float f) {
  unsigned u = __float_as_uint(f);
  u += 0x7FFFu + ((u >> 16) & 1u);       // round-to-nearest-even
  return (unsigned short)(u >> 16);
}
__device__ __forceinline__ float bf2f(unsigned short h) {
  return __uint_as_float(((unsigned)h) << 16);
}
__device__ __forceinline__ unsigned long long shfl_xor_u64(unsigned long long v, int m) {
  unsigned lo = (unsigned)v, hi = (unsigned)(v >> 32);
  lo = __shfl_xor(lo, m, 32);
  hi = __shfl_xor(hi, m, 32);
  return ((unsigned long long)hi << 32) | lo;
}

// ---------------- elementwise fp32 -> bf16 ----------------
__global__ void cvt_bf16_kernel(const float* __restrict__ src,
                                unsigned short* __restrict__ dst, size_t n) {
  size_t i = (size_t)blockIdx.x * blockDim.x + threadIdx.x;
  if (i < n) dst[i] = f2bf(src[i]);
}

// ---------------- row squared-norms ----------------
__global__ void rownorm_kernel(const float* __restrict__ src,
                               float* __restrict__ dst, int ncols) {
  __shared__ float red[256];
  const int row = blockIdx.x;
  float s = 0.f;
  for (int d = threadIdx.x; d < ncols; d += 256) {
    float v = src[(size_t)row * ncols + d];
    s += v * v;
  }
  red[threadIdx.x] = s; __syncthreads();
  for (int st = 128; st > 0; st >>= 1) {
    if (threadIdx.x < st) red[threadIdx.x] += red[threadIdx.x + st];
    __syncthreads();
  }
  if (threadIdx.x == 0) dst[row] = red[0];
}

__global__ void initkeys_kernel(unsigned long long* __restrict__ keys, int n) {
  int i = blockIdx.x * blockDim.x + threadIdx.x;
  if (i < n) keys[i] = ~0ull;
}

// ---------------- KNN: bf16 WMMA GEMM + streaming argmin ----------------
// grid (49, 16), block (32,4). Wave w handles 512 memory rows (32 n-tiles).
__global__ void knn_kernel(const unsigned short* __restrict__ qbf_us,
                           const unsigned short* __restrict__ mbf_us,
                           const float* __restrict__ qq,
                           const float* __restrict__ mm,
                           unsigned long long* __restrict__ keys) {
  const __bf16* qbf = (const __bf16*)qbf_us;
  const __bf16* mbf = (const __bf16*)mbf_us;
  const int lane = threadIdx.x;
  const int half = lane >> 4;
  const int lsub = lane & 15;
  const int qt   = blockIdx.x;                       // query tile (16 rows)
  const int wid  = blockIdx.y * 4 + threadIdx.y;     // 0..63
  const int n0   = wid * 512;

  // A operand: this lane carries query row qt*16+lsub (K split by half)
  const __bf16* aRow = qbf + (size_t)(qt * 16 + lsub) * DQ;

  float qv[8];
  #pragma unroll
  for (int r = 0; r < 8; ++r) qv[r] = qq[qt * 16 + half * 8 + r];

  unsigned long long best[8];
  #pragma unroll
  for (int r = 0; r < 8; ++r) best[r] = ~0ull;

  for (int t = 0; t < 32; ++t) {
    const int col = n0 + t * 16 + lsub;              // memory row = B column
    const __bf16* bRow = mbf + (size_t)col * DQ;
    if (t + 1 < 32)                                  // -> global_prefetch_b8
      __builtin_prefetch(mbf + (size_t)(n0 + (t + 1) * 16 + lsub) * DQ, 0, 1);

    v8f c = {};
    #pragma unroll 4
    for (int kc = 0; kc < DQ; kc += 32) {
      bf16x8 a0 = *(const bf16x8*)(aRow + kc + half * 8);
      bf16x8 a1 = *(const bf16x8*)(aRow + kc + 16 + half * 8);
      v16bf a = __builtin_shufflevector(a0, a1, 0,1,2,3,4,5,6,7,8,9,10,11,12,13,14,15);
      bf16x8 b0 = *(const bf16x8*)(bRow + kc + half * 8);
      bf16x8 b1 = *(const bf16x8*)(bRow + kc + 16 + half * 8);
      v16bf b = __builtin_shufflevector(b0, b1, 0,1,2,3,4,5,6,7,8,9,10,11,12,13,14,15);
      c = __builtin_amdgcn_wmma_f32_16x16x32_bf16(false, a, false, b,
                                                  (short)0, c, false, false);
    }
    const float mmv = mm[col];
    #pragma unroll
    for (int r = 0; r < 8; ++r) {
      float d = qv[r] + mmv - 2.0f * c[r];           // sqd
      d = d > 0.f ? d : 0.f;
      unsigned long long key =
          ((unsigned long long)__float_as_uint(d) << 32) | (unsigned)col;
      if (key < best[r]) best[r] = key;
    }
  }
  // min across the 16 lanes of each half (N dimension)
  #pragma unroll
  for (int r = 0; r < 8; ++r) {
    unsigned long long k = best[r];
    for (int m = 1; m < 16; m <<= 1) {
      unsigned long long o = shfl_xor_u64(k, m);
      if (o < k) k = o;
    }
    if (lsub == 0) atomicMin(&keys[qt * 16 + half * 8 + r], k);
  }
}

// ---------------- gather neighbor, exact fp32 dist, build combined ----------------
__global__ void gather_kernel(const float* __restrict__ q,
                              const float* __restrict__ mem,
                              const unsigned short* __restrict__ qbf,
                              const unsigned short* __restrict__ mbf,
                              const unsigned long long* __restrict__ keys,
                              unsigned short* __restrict__ comb,
                              float* __restrict__ outDist) {
  __shared__ float red[128];
  __shared__ int sIdx;
  const int m = blockIdx.x, t = threadIdx.x;
  if (t == 0) sIdx = (int)(keys[m] & 0xffffffffu);
  __syncthreads();
  const int idx = sIdx;
  float s = 0.f;
  for (int d = t; d < DQ; d += 128) {
    float df = q[(size_t)m * DQ + d] - mem[(size_t)idx * DQ + d];
    s += df * df;
  }
  red[t] = s; __syncthreads();
  for (int st = 64; st > 0; st >>= 1) {
    if (t < st) red[t] += red[t + st];
    __syncthreads();
  }
  if (t == 0) outDist[m] = red[0];
  for (int d = t; d < DQ; d += 128) {
    comb[(size_t)m * EMB + d]      = qbf[(size_t)m * DQ + d];
    comb[(size_t)m * EMB + DQ + d] = mbf[(size_t)idx * DQ + d];
  }
}

// ---------------- weight repack: OIHW fp32 -> [kk][Cout][Cin] bf16 ----------------
__global__ void packw_kernel(const float* __restrict__ w,
                             unsigned short* __restrict__ wp,
                             int Cout, int Cin) {
  int tid = blockIdx.x * blockDim.x + threadIdx.x;
  int total = Cout * Cin * 9;
  if (tid >= total) return;
  int oc  = tid / (Cin * 9);
  int rem = tid % (Cin * 9);
  int ic  = rem / 9;
  int kk  = rem % 9;
  wp[((size_t)kk * Cout + oc) * Cin + ic] = f2bf(w[tid]);
}

// ---------------- 3x3 SAME conv, implicit GEMM, bf16 WMMA ----------------
// in: [H][W][Cin] bf16; wp: [9][Cout][Cin] bf16; out: [H][W][Cout] bf16 (+bias,+relu)
// grid (H*W/16, ceil(Cout/64)), block (32,4): one 16oc x 16px tile per wave.
__global__ void conv3x3_wmma_kernel(const unsigned short* __restrict__ in_us,
                                    const unsigned short* __restrict__ wp_us,
                                    const float* __restrict__ bias,
                                    unsigned short* __restrict__ out,
                                    int H, int W, int Cin, int Cout, int relu) {
  const __bf16* in = (const __bf16*)in_us;
  const __bf16* wp = (const __bf16*)wp_us;
  const int lane = threadIdx.x, half = lane >> 4, lsub = lane & 15;
  const int pxt = blockIdx.x;
  const int oct = blockIdx.y * blockDim.y + threadIdx.y;
  if (oct * 16 >= Cout) return;                      // wave-uniform

  const int n = pxt * 16 + lsub;                     // B pixel for this lane
  const int y = n / W, x = n % W;
  const int ocA = oct * 16 + lsub;                   // A row for this lane

  v8f c = {};
  for (int kk = 0; kk < 9; ++kk) {
    const int dy = kk / 3 - 1, dx = kk % 3 - 1;
    const int yy = y + dy, xx = x + dx;
    const bool valid = (yy >= 0) && (yy < H) && (xx >= 0) && (xx < W);
    const __bf16* aBase = wp + ((size_t)kk * Cout + ocA) * Cin;
    const __bf16* bBase = in + (size_t)(valid ? (yy * W + xx) : 0) * Cin;
    for (int kc = 0; kc < Cin; kc += 32) {
      bf16x8 a0 = *(const bf16x8*)(aBase + kc + half * 8);
      bf16x8 a1 = *(const bf16x8*)(aBase + kc + 16 + half * 8);
      v16bf a = __builtin_shufflevector(a0, a1, 0,1,2,3,4,5,6,7,8,9,10,11,12,13,14,15);
      bf16x8 b0 = {}, b1 = {};
      if (valid) {
        b0 = *(const bf16x8*)(bBase + kc + half * 8);
        b1 = *(const bf16x8*)(bBase + kc + 16 + half * 8);
      }
      v16bf b = __builtin_shufflevector(b0, b1, 0,1,2,3,4,5,6,7,8,9,10,11,12,13,14,15);
      c = __builtin_amdgcn_wmma_f32_16x16x32_bf16(false, a, false, b,
                                                  (short)0, c, false, false);
    }
  }
  const int ocOut = oct * 16 + half * 8;             // C/D: M = half*8 + r
  us8 o;
  #pragma unroll
  for (int r = 0; r < 8; ++r) {
    float v = c[r] + bias[ocOut + r];
    if (relu) v = v > 0.f ? v : 0.f;
    o[r] = f2bf(v);
  }
  *(us8*)(out + (size_t)n * Cout + ocOut) = o;       // one b128 store
}

// ---------------- 2x bilinear upsample, half-pixel centers, edge clamp ----------------
__global__ void up2x_kernel(const unsigned short* __restrict__ in,
                            unsigned short* __restrict__ out,
                            int H, int W, int C) {
  int tid = blockIdx.x * blockDim.x + threadIdx.x;
  int total = 4 * H * W * C;
  if (tid >= total) return;
  int c = tid % C;
  int rem = tid / C;
  int ox = rem % (2 * W);
  int oy = rem / (2 * W);
  float sy = 0.5f * oy - 0.25f, sx = 0.5f * ox - 0.25f;
  int y0 = (int)floorf(sy), x0 = (int)floorf(sx);
  float wy = sy - (float)y0, wx = sx - (float)x0;
  int y0c = min(max(y0, 0), H - 1), y1c = min(max(y0 + 1, 0), H - 1);
  int x0c = min(max(x0, 0), W - 1), x1c = min(max(x0 + 1, 0), W - 1);
  float v00 = bf2f(in[((size_t)y0c * W + x0c) * C + c]);
  float v01 = bf2f(in[((size_t)y0c * W + x1c) * C + c]);
  float v10 = bf2f(in[((size_t)y1c * W + x0c) * C + c]);
  float v11 = bf2f(in[((size_t)y1c * W + x1c) * C + c]);
  float v = (1.f - wy) * ((1.f - wx) * v00 + wx * v01) +
            wy         * ((1.f - wx) * v10 + wx * v11);
  out[((size_t)oy * (2 * W) + ox) * C + c] = f2bf(v);
}

// ---------------- head conv 288 -> 1, direct fp32 ----------------
__global__ void head_kernel(const unsigned short* __restrict__ in,  // [224][224][288]
                            const float* __restrict__ wh,
                            const float* __restrict__ bh,
                            float* __restrict__ score) {
  int px = blockIdx.x * blockDim.x + threadIdx.x;
  if (px >= 224 * 224) return;
  int y = px / 224, x = px % 224;
  float acc = bh[0];
  for (int ic = 0; ic < 288; ++ic) {
    const float* w9 = wh + ic * 9;
    #pragma unroll
    for (int kk = 0; kk < 9; ++kk) {
      int dy = kk / 3 - 1, dx = kk % 3 - 1;
      int yy = y + dy, xx = x + dx;
      if (yy >= 0 && yy < 224 && xx >= 0 && xx < 224)
        acc += w9[kk] * bf2f(in[((size_t)yy * 224 + xx) * 288 + ic]);
    }
  }
  score[px] = acc;
}

// ---------------- launch ----------------
extern "C" void kernel_launch(void* const* d_in, const int* in_sizes, int n_in,
                              void* d_out, int out_size, void* d_ws, size_t ws_size,
                              hipStream_t stream) {
  const float* query = (const float*)d_in[0];
  const float* membk = (const float*)d_in[1];
  const float* w1 = (const float*)d_in[2]; const float* b1 = (const float*)d_in[3];
  const float* w2 = (const float*)d_in[4]; const float* b2 = (const float*)d_in[5];
  const float* w3 = (const float*)d_in[6]; const float* b3 = (const float*)d_in[7];
  const float* wh = (const float*)d_in[8]; const float* bh = (const float*)d_in[9];

  float* score   = (float*)d_out;           // [224*224]
  float* outDist = score + 224 * 224;       // [784]

  // carve workspace (256B aligned slices)
  char* p = (char*)d_ws;
  auto carve = [&](size_t bytes) -> void* {
    void* r = (void*)p;
    p += (bytes + 255) & ~(size_t)255;
    return r;
  };
  unsigned short* qbf = (unsigned short*)carve((size_t)NP * DQ * 2);
  unsigned short* mbf = (unsigned short*)carve((size_t)MBK * DQ * 2);
  float* qq = (float*)carve((size_t)NP * 4);
  float* mm = (float*)carve((size_t)MBK * 4);
  unsigned long long* keys = (unsigned long long*)carve((size_t)NP * 8);
  unsigned short* comb = (unsigned short*)carve((size_t)NP * EMB * 2);
  unsigned short* wp1 = (unsigned short*)carve((size_t)9 * 1152 * 2304 * 2);
  unsigned short* wp2 = (unsigned short*)carve((size_t)9 * 576 * 1152 * 2);
  unsigned short* wp3 = (unsigned short*)carve((size_t)9 * 288 * 576 * 2);
  unsigned short* x1  = (unsigned short*)carve((size_t)784 * 1152 * 2);
  unsigned short* up1 = (unsigned short*)carve((size_t)3136 * 1152 * 2);
  unsigned short* x2  = (unsigned short*)carve((size_t)3136 * 576 * 2);
  unsigned short* up2 = (unsigned short*)carve((size_t)12544 * 576 * 2);
  unsigned short* x3  = (unsigned short*)carve((size_t)12544 * 288 * 2);
  unsigned short* up3 = (unsigned short*)carve((size_t)50176 * 288 * 2);

  // 1) bf16 conversions
  {
    size_t nq = (size_t)NP * DQ, nm = (size_t)MBK * DQ;
    cvt_bf16_kernel<<<(unsigned)((nq + 255) / 256), 256, 0, stream>>>(query, qbf, nq);
    cvt_bf16_kernel<<<(unsigned)((nm + 255) / 256), 256, 0, stream>>>(membk, mbf, nm);
  }
  // 2) row norms
  rownorm_kernel<<<NP, 256, 0, stream>>>(query, qq, DQ);
  rownorm_kernel<<<MBK, 256, 0, stream>>>(membk, mm, DQ);
  // 3) argmin keys init + KNN WMMA GEMM
  initkeys_kernel<<<(NP + 255) / 256, 256, 0, stream>>>(keys, NP);
  knn_kernel<<<dim3(49, 16), dim3(32, 4), 0, stream>>>(qbf, mbf, qq, mm, keys);
  // 4) gather neighbor + exact dist + combined (NHWC = [784][2304])
  gather_kernel<<<NP, 128, 0, stream>>>(query, membk, qbf, mbf, keys, comb, outDist);
  // 5) weight repack to [kk][Cout][Cin] bf16
  packw_kernel<<<(1152 * 2304 * 9 + 255) / 256, 256, 0, stream>>>(w1, wp1, 1152, 2304);
  packw_kernel<<<(576 * 1152 * 9 + 255) / 256, 256, 0, stream>>>(w2, wp2, 576, 1152);
  packw_kernel<<<(288 * 576 * 9 + 255) / 256, 256, 0, stream>>>(w3, wp3, 288, 576);
  // 6) decoder: conv -> up2x -> conv -> up2x -> conv -> up2x -> head
  conv3x3_wmma_kernel<<<dim3(49, 18), dim3(32, 4), 0, stream>>>(
      comb, wp1, b1, x1, 28, 28, 2304, 1152, 1);
  up2x_kernel<<<(4 * 28 * 28 * 1152 + 255) / 256, 256, 0, stream>>>(x1, up1, 28, 28, 1152);
  conv3x3_wmma_kernel<<<dim3(196, 9), dim3(32, 4), 0, stream>>>(
      up1, wp2, b2, x2, 56, 56, 1152, 576, 1);
  up2x_kernel<<<(4 * 56 * 56 * 576 + 255) / 256, 256, 0, stream>>>(x2, up2, 56, 56, 576);
  conv3x3_wmma_kernel<<<dim3(784, 5), dim3(32, 4), 0, stream>>>(
      up2, wp3, b3, x3, 112, 112, 576, 288, 1);
  up2x_kernel<<<(4 * 112 * 112 * 288 + 255) / 256, 256, 0, stream>>>(x3, up3, 112, 112, 288);
  head_kernel<<<(224 * 224 + 255) / 256, 256, 0, stream>>>(up3, wh, bh, score);
}
